// HummingLayer_46024869544571
// MI455X (gfx1250) — matrix operand order repaired and checked by reading, not compiled, not run on hardware
//
#include <hip/hip_runtime.h>

typedef __attribute__((ext_vector_type(16))) _Float16 v16h;
typedef __attribute__((ext_vector_type(8)))  _Float16 v8h;
typedef __attribute__((ext_vector_type(4)))  _Float16 v4h;
typedef __attribute__((ext_vector_type(8)))  float    v8f;
typedef __attribute__((ext_vector_type(2)))  unsigned int v2u;

#define M_DIM 16
#define N_DIM 8192
#define K_DIM 8192
#define GS_Q  128
#define G_NUM (K_DIM / GS_Q)                 /* 64 K-groups           */
#define WAVES_PER_TILE 8                     /* split-K factor        */
#define GROUPS_PER_WAVE (G_NUM / WAVES_PER_TILE)   /* 8 */

#define A16_WS_BYTES ((size_t)M_DIM * K_DIM * 2)   /* 256 KB f16 acts  */

// ---------------------------------------------------------------------------
// Pre-pass 1: convert fp32 activations [16, K] -> f16 in workspace.
// ---------------------------------------------------------------------------
__global__ void cvt_a_kernel(const float4* __restrict__ in, _Float16* __restrict__ out) {
    const int i = blockIdx.x * blockDim.x + threadIdx.x;       // float4 index
    const float4 v = in[i];
    v4h h;
    h[0] = (_Float16)v.x; h[1] = (_Float16)v.y;
    h[2] = (_Float16)v.z; h[3] = (_Float16)v.w;
    *(v4h*)(out + (size_t)i * 4) = h;
}

// ---------------------------------------------------------------------------
// Pre-pass 2: per-(group, row) activation sums  asum[g][m] = sum_{k in g} a16.
// 64*16 = 1024 outputs, each a 128-element sum. Used to fold zp+bias out of
// the inner loop:  sum a*(q-zp)*sc = sc*sum a*(1024+q) - sc*(1024+zp)*asum.
// ---------------------------------------------------------------------------
__global__ void asum_kernel(const _Float16* __restrict__ a16, float* __restrict__ asum) {
    const int t = blockIdx.x * blockDim.x + threadIdx.x;   // 0..1023
    const int g = t >> 4;
    const int m = t & 15;
    const _Float16* p = a16 + (size_t)m * K_DIM + (size_t)g * GS_Q;
    float s = 0.f;
    #pragma unroll
    for (int j = 0; j < GS_Q; j += 8) {
        const v8h v = *(const v8h*)(p + j);
        #pragma unroll
        for (int e = 0; e < 8; ++e) s += (float)v[e];
    }
    asum[(g << 4) + m] = s;
}

// ---------------------------------------------------------------------------
// Nibble -> f16 bit trick:  as_f16(0x6400 | q) == 1024 + q  (exact).
// One v_perm_b32 + one v_or_b32 per packed f16 pair; no v_cvt at all.
// ---------------------------------------------------------------------------
__device__ __forceinline__ void unpack8_biased(unsigned d, unsigned* out) {
    const unsigned ev = d & 0x0F0F0F0Fu;          // nibbles 0,2,4,6 in bytes
    const unsigned od = (d >> 4) & 0x0F0F0F0Fu;   // nibbles 1,3,5,7 in bytes
    // perm sel: result byte0 = ev.byte_j (idx 4+j), byte1 = 0x00 (idx 12),
    //           byte2 = od.byte_j (idx j),  byte3 = 0x00 (idx 12)
    out[0] = __builtin_amdgcn_perm(ev, od, 0x0C000C04u) | 0x64006400u;
    out[1] = __builtin_amdgcn_perm(ev, od, 0x0C010C05u) | 0x64006400u;
    out[2] = __builtin_amdgcn_perm(ev, od, 0x0C020C06u) | 0x64006400u;
    out[3] = __builtin_amdgcn_perm(ev, od, 0x0C030C07u) | 0x64006400u;
}

// ---------------------------------------------------------------------------
// Main kernel: one block (8 waves) per 16-column N tile; each wave owns 8 of
// the 64 K-groups (split-K), accumulating with v_wmma_f32_16x16x32_f16 on
// bias-encoded weights (1024+q); per-group algebraic correction applies
// scale and removes the (1024+zp) term using the precomputed row sums.
// ---------------------------------------------------------------------------
__global__ void __launch_bounds__(32 * WAVES_PER_TILE)
w4a16_wmma_gemm_kernel(const _Float16* __restrict__ A,    // [16, K] f16 (ws)
                       const float*    __restrict__ Asum, // [G, 16] (ws)
                       const int*      __restrict__ Wq,   // [N, K/8] packed int4
                       const float*    __restrict__ Sc,   // [G, N]
                       const int*      __restrict__ Zp,   // [G, N/8] packed int4
                       const float*    __restrict__ Bias, // [N]
                       const float*    __restrict__ Gsc,  // [1]
                       float*          __restrict__ Out)  // [16, N]
{
    __shared__ float red[WAVES_PER_TILE][32][8];

    const int lane  = threadIdx.x & 31;
    const int wave  = threadIdx.x >> 5;
    const int n     = blockIdx.x * 16 + (lane & 15);   // this lane's output col
    const int mrow  = lane & 15;                       // A-fragment row
    const int koffA = (lane < 16) ? 0 : 16;            // A lane K sub-offset
    const int koffW = (lane < 16) ? 0 : 16;            // B lane K sub-offset
    const int m0    = (lane >> 4) << 3;                // C rows: 0..7 or 8..15
    const int zsh   = (n & 7) * 4;

    const _Float16*      arow = A + (size_t)mrow * K_DIM;
    const unsigned char* wrow = (const unsigned char*)(Wq + (size_t)n * (K_DIM / 8));

    v8f total = {};

    for (int gi = 0; gi < GROUPS_PER_WAVE; ++gi) {
        const int g     = wave * GROUPS_PER_WAVE + gi;
        const int kbase = g * GS_Q;

        // Prefetch next group's weight bytes for this lane's row.
        if (gi + 1 < GROUPS_PER_WAVE)
            __builtin_prefetch(wrow + ((kbase + GS_Q + koffW) >> 1), 0, 0);

        const float    sc  = Sc[(size_t)g * N_DIM + n];
        const unsigned zdw = (unsigned)Zp[(size_t)g * (N_DIM / 8) + (n >> 3)];
        const float    zpf = (float)((zdw >> zsh) & 0xF);

        v8f gacc = {};
        #pragma unroll
        for (int kk = 0; kk < GS_Q; kk += 32) {
            const int kc = kbase + kk;

            // A fragment (f16, WMMA lane layout): two aligned b128 loads.
            const v8h a0 = *(const v8h*)(arow + kc + koffA);
            const v8h a1 = *(const v8h*)(arow + kc + koffA + 8);
            v16h a;
            #pragma unroll
            for (int i = 0; i < 8; ++i) { a[i] = a0[i]; a[8 + i] = a1[i]; }

            // B fragment: 16 nibbles = one aligned b64 load, bit-trick expand.
            const v2u w = *(const v2u*)(wrow + ((kc + koffW) >> 1));
            union { v16h h; unsigned u[8]; } B;
            unpack8_biased(w.x, B.u);
            unpack8_biased(w.y, B.u + 4);

            gacc = __builtin_amdgcn_wmma_f32_16x16x32_f16(
                       /*neg_a=*/false, a, /*neg_b=*/false, B.h,
                       /*c_mod=*/(short)0, gacc,
                       /*reuse_a=*/false, /*reuse_b=*/false);
        }

        // Per-group correction: sc * gacc - sc*(1024+zp) * rowsum(a).
        const float  corr = sc * (1024.0f + zpf);
        const float* as   = Asum + (g << 4) + m0;
        #pragma unroll
        for (int i = 0; i < 8; ++i)
            total[i] += sc * gacc[i] - corr * as[i];
    }

    // Split-K reduction across the 8 waves through LDS.
    #pragma unroll
    for (int i = 0; i < 8; ++i) red[wave][lane][i] = total[i];
    __syncthreads();

    if (wave == 0) {
        const float gs = Gsc[0];
        const float bv = Bias[n];
        #pragma unroll
        for (int i = 0; i < 8; ++i) {
            float s = 0.f;
            #pragma unroll
            for (int wv = 0; wv < WAVES_PER_TILE; ++wv) s += red[wv][lane][i];
            Out[(size_t)(m0 + i) * N_DIM + n] = fmaf(s, gs, bv);
        }
    }
}

// ---------------------------------------------------------------------------
extern "C" void kernel_launch(void* const* d_in, const int* in_sizes, int n_in,
                              void* d_out, int out_size, void* d_ws, size_t ws_size,
                              hipStream_t stream) {
    const float* inputs  = (const float*)d_in[0];   // [16, 8192] f32
    const int*   weight  = (const int*)  d_in[1];   // [8192, 1024] packed int4
    const float* wscale  = (const float*)d_in[2];   // [64, 8192] f32
    const int*   zpoint  = (const int*)  d_in[3];   // [64, 1024] packed int4
    const float* bias    = (const float*)d_in[4];   // [8192] f32
    const float* gscale  = (const float*)d_in[5];   // [1] f32
    float*       out     = (float*)d_out;           // [16, 8192] f32

    _Float16* a16  = (_Float16*)d_ws;                              // 256 KB
    float*    asum = (float*)((char*)d_ws + A16_WS_BYTES);         // 4 KB

    // 1) fp32 -> f16 activation conversion (131072 elems, 4 per thread).
    const int cvt_threads = (M_DIM * K_DIM) / 4;    // 32768
    cvt_a_kernel<<<cvt_threads / 256, 256, 0, stream>>>((const float4*)inputs, a16);

    // 2) Per-(group, row) activation sums (1024 outputs).
    asum_kernel<<<(G_NUM * M_DIM) / 256, 256, 0, stream>>>(a16, asum);

    // 3) WMMA dequant-GEMM: 512 tiles of 16 columns, 8 waves (split-K) each.
    w4a16_wmma_gemm_kernel<<<N_DIM / 16, 32 * WAVES_PER_TILE, 0, stream>>>(
        a16, asum, weight, wscale, zpoint, bias, gscale, out);
}
